// SimpleLSTM_82145544503557
// MI455X (gfx1250) — compile-verified, matrix operand order checked by reading, not run on hardware
//
#include <hip/hip_runtime.h>
#include <math.h>

typedef __attribute__((ext_vector_type(16))) _Float16 v16h;
typedef __attribute__((ext_vector_type(8)))  float    v8f;

#define Tn 256
#define Fn 32
#define Hn 64
#define MROWS 16          // batch rows per workgroup
#define NTHREADS 128      // 4 wave32 waves
#define NFRAG 112         // 16 (W1) + 32 (U1) + 32 (W2) + 32 (U2) B-fragments
#define SMEM_BYTES (114688 + 1024 + 2048 + 2048 + 4608 + 4608) // 129024

#define WM(A,B,C) __builtin_amdgcn_wmma_f32_16x16x32_f16(false,(A),false,(B),(short)0,(C),false,false)

// Branch-free activations: v_max/v_min clamp + v_exp_f32 + v_rcp_f32.
// Straight-line VALU co-executes with the WMMA (XDL) pipeline; no EXEC
// divergence between wmma instructions.
__device__ __forceinline__ float fast_sig(float v) {
    v = fminf(fmaxf(v, -30.f), 30.f);
    return __builtin_amdgcn_rcpf(1.f + __expf(-v));
}
__device__ __forceinline__ float fast_tanh(float v) {
    v = fminf(fmaxf(v, -15.f), 15.f);
    const float e = __expf(2.f * v);
    return (e - 1.f) * __builtin_amdgcn_rcpf(e + 1.f);
}

// B-fragment element mapping (16-bit, 16x16x32): lane l holds col n=l&15;
// element e maps to K = (e&7) + ((e>>3)<<4) + ((l>>4)<<3). A-fragments mirror
// this with row m = l&15. Weights/h are stored pre-swizzled so each lane's 16
// halfs are one contiguous 32B LDS read (ds_load_b128 x2).

__global__ __launch_bounds__(NTHREADS) void lstm2_fused_wmma(
    const float* __restrict__ x,
    const float* __restrict__ W1, const float* __restrict__ U1, const float* __restrict__ b1,
    const float* __restrict__ W2, const float* __restrict__ U2, const float* __restrict__ b2,
    const float* __restrict__ Wd, const float* __restrict__ bd,
    const float* __restrict__ Wo, const float* __restrict__ bo,
    float* __restrict__ out)
{
    extern __shared__ char smem[];
    _Float16* wB   = (_Float16*)smem;        // 112 frags * 512 halfs = 114688 B
    _Float16* xA   = wB  + NFRAG*512;        // 512 halfs: x_t tile, A-layout
    _Float16* h1A  = xA  + 512;              // 2 chunks * 512 halfs
    _Float16* h2A  = h1A + 1024;             // 2 chunks * 512 halfs
    float*    hfin = (float*)(h2A + 1024);   // [16][72] final h2 (padded)
    float*    dbuf = hfin + MROWS*72;        // [16][72] dense intermediate

    const int tid  = threadIdx.x;
    const int w    = tid >> 5;               // wave id 0..3
    const int lane = tid & 31;
    const int n    = lane & 15;
    const int hi   = lane >> 4;              // 0/1 (row-half selector in C/D layout)
    const int b0   = blockIdx.x * MROWS;

    // ---- one-time: convert weights f32->f16 into pre-swizzled B fragments ----
    for (int idx = tid; idx < NFRAG*512; idx += NTHREADS) {
        const int fi = idx >> 9;
        const int l  = (idx >> 4) & 31;
        const int e  = idx & 15;
        const int kin = (e & 7) + ((e >> 3) << 4) + ((l >> 4) << 3); // K within 32-chunk
        const int col16 = l & 15;
        const float* mat; int tile, row;
        if (fi < 16)      { mat = W1; tile = fi;            row = kin; }
        else if (fi < 48) { mat = U1; tile = (fi-16) >> 1;  row = ((fi-16)&1)*32 + kin; }
        else if (fi < 80) { mat = W2; tile = (fi-48) >> 1;  row = ((fi-48)&1)*32 + kin; }
        else              { mat = U2; tile = (fi-80) >> 1;  row = ((fi-80)&1)*32 + kin; }
        wB[idx] = (_Float16)mat[row*256 + tile*16 + col16];
    }
    for (int i = tid; i < 1024; i += NTHREADS) { h1A[i] = (_Float16)0.f; h2A[i] = (_Float16)0.f; }

    // biases for this thread's unit column u = 16w + n (Keras gate order i,f,g,o)
    const int u = 16*w + n;
    const float bi1=b1[u], bf1=b1[64+u], bg1=b1[128+u], bo1=b1[192+u];
    const float bi2=b2[u], bf2=b2[64+u], bg2=b2[128+u], bo2=b2[192+u];

    // h write-back constants: u -> A-layout (chunk, lane', elem)
    const int kk    = ((w & 1) << 4) | n;     // u & 31
    const int chunk = w >> 1;                 // u >> 5
    const int hsel  = (((kk >> 3) & 1) << 4); // lane-group select
    const int ep    = (kk & 7) + ((kk >> 4) << 3);
    const int hbase = chunk*512 + hsel*16 + ep;

    // x staging: this thread loads 4 contiguous f32 of row m_x each step
    const int m_x = tid >> 3;
    const int f0  = (tid & 7) * 4;
    const float* xrow = x + ((size_t)(b0 + m_x) * Tn) * Fn + f0;
    int xdst[4];
    #pragma unroll
    for (int j = 0; j < 4; ++j) {
        const int kx = f0 + j;
        const int lp = m_x + (((kx >> 3) & 1) << 4);
        xdst[j] = lp*16 + (kx & 7) + ((kx >> 4) << 3);
    }

    v8f c1 = {}; v8f c2 = {};

    for (int t = 0; t < Tn; ++t) {
        // stage x_t tile into A-layout LDS
        const float4 xv = *(const float4*)(xrow + t*Fn);
        xA[xdst[0]] = (_Float16)xv.x; xA[xdst[1]] = (_Float16)xv.y;
        xA[xdst[2]] = (_Float16)xv.z; xA[xdst[3]] = (_Float16)xv.w;
        if (t + 1 < Tn) __builtin_prefetch(xrow + (t+1)*Fn, 0, 3);
        __syncthreads();                                   // xA, h1A(t-1), h2A(t-1) ready

        // ---- layer 1: z1 = x@W1 + h1@U1 ----
        v16h ax  = *(const v16h*)(xA  + lane*16);
        v16h a10 = *(const v16h*)(h1A + lane*16);
        v16h a11 = *(const v16h*)(h1A + 512 + lane*16);
        v8f acc[4] = {{},{},{},{}};
        #pragma unroll
        for (int q = 0; q < 4; ++q) {                      // q: i,f,g,o
            const int tile = 4*q + w;
            v16h bw  = *(const v16h*)(wB + (size_t)tile*512              + lane*16);
            acc[q] = WM(ax,  bw,  acc[q]);
            v16h bu0 = *(const v16h*)(wB + (size_t)(16 + tile*2 + 0)*512 + lane*16);
            acc[q] = WM(a10, bu0, acc[q]);
            v16h bu1 = *(const v16h*)(wB + (size_t)(16 + tile*2 + 1)*512 + lane*16);
            acc[q] = WM(a11, bu1, acc[q]);
        }
        __syncthreads();                                   // all reads of h1A(t-1) done

        #pragma unroll
        for (int r = 0; r < 8; ++r) {                      // cell update, register-local
            const float ig = fast_sig(acc[0][r] + bi1);
            const float fg = fast_sig(acc[1][r] + bf1);
            const float gg = fast_tanh(acc[2][r] + bg1);
            const float og = fast_sig(acc[3][r] + bo1);
            c1[r] = fg * c1[r] + ig * gg;
            const float h = og * fast_tanh(c1[r]);
            h1A[hbase + (r + hi*8)*16] = (_Float16)h;      // write A-layout h1(t)
        }
        __syncthreads();                                   // h1A(t) visible

        // ---- layer 2: z2 = h1(t)@W2 + h2(t-1)@U2 ----
        v16h a1n0 = *(const v16h*)(h1A + lane*16);
        v16h a1n1 = *(const v16h*)(h1A + 512 + lane*16);
        v16h a20  = *(const v16h*)(h2A + lane*16);
        v16h a21  = *(const v16h*)(h2A + 512 + lane*16);
        #pragma unroll
        for (int q = 0; q < 4; ++q) {
            const int tile = 4*q + w;
            acc[q] = (v8f){};
            v16h bw0 = *(const v16h*)(wB + (size_t)(48 + tile*2 + 0)*512 + lane*16);
            acc[q] = WM(a1n0, bw0, acc[q]);
            v16h bw1 = *(const v16h*)(wB + (size_t)(48 + tile*2 + 1)*512 + lane*16);
            acc[q] = WM(a1n1, bw1, acc[q]);
            v16h bu0 = *(const v16h*)(wB + (size_t)(80 + tile*2 + 0)*512 + lane*16);
            acc[q] = WM(a20,  bu0, acc[q]);
            v16h bu1 = *(const v16h*)(wB + (size_t)(80 + tile*2 + 1)*512 + lane*16);
            acc[q] = WM(a21,  bu1, acc[q]);
        }
        __syncthreads();                                   // all reads of h2A(t-1) done

        #pragma unroll
        for (int r = 0; r < 8; ++r) {
            const float ig = fast_sig(acc[0][r] + bi2);
            const float fg = fast_sig(acc[1][r] + bf2);
            const float gg = fast_tanh(acc[2][r] + bg2);
            const float og = fast_sig(acc[3][r] + bo2);
            c2[r] = fg * c2[r] + ig * gg;
            const float h = og * fast_tanh(c2[r]);
            h2A[hbase + (r + hi*8)*16] = (_Float16)h;
            if (t == Tn - 1) hfin[(r + hi*8)*72 + u] = h;  // f32 copy for dense head
        }
    }
    __syncthreads();

    // ---- Dense(64, sigmoid) ----
    for (int o = tid; o < MROWS*Hn; o += NTHREADS) {
        const int m = o >> 6, uc = o & 63;
        float a = bd[uc];
        for (int k = 0; k < Hn; ++k) a += hfin[m*72 + k] * Wd[k*64 + uc];
        dbuf[m*72 + uc] = fast_sig(a);
    }
    __syncthreads();

    // ---- Dense(1, sigmoid) ----
    if (tid < MROWS) {
        float a = bo[0];
        for (int k = 0; k < Hn; ++k) a += dbuf[tid*72 + k] * Wo[k];
        out[b0 + tid] = fast_sig(a);
    }
}

extern "C" void kernel_launch(void* const* d_in, const int* in_sizes, int n_in,
                              void* d_out, int out_size, void* d_ws, size_t ws_size,
                              hipStream_t stream) {
    (void)n_in; (void)d_ws; (void)ws_size; (void)out_size;
    const float* x  = (const float*)d_in[0];
    const float* W1 = (const float*)d_in[1];
    const float* U1 = (const float*)d_in[2];
    const float* b1 = (const float*)d_in[3];
    const float* W2 = (const float*)d_in[4];
    const float* U2 = (const float*)d_in[5];
    const float* b2 = (const float*)d_in[6];
    const float* Wd = (const float*)d_in[7];
    const float* bd = (const float*)d_in[8];
    const float* Wo = (const float*)d_in[9];
    const float* bo = (const float*)d_in[10];
    float* out = (float*)d_out;

    const int B = in_sizes[0] / (Tn * Fn);   // 4096
    hipFuncSetAttribute((const void*)lstm2_fused_wmma,
                        hipFuncAttributeMaxDynamicSharedMemorySize, SMEM_BYTES);
    lstm2_fused_wmma<<<B / MROWS, NTHREADS, SMEM_BYTES, stream>>>(
        x, W1, U1, b1, W2, U2, b2, Wd, bd, Wo, bo, out);
}
